// SwinLayer_66383014527691
// MI455X (gfx1250) — matrix operand order repaired and checked by reading, not compiled, not run on hardware
//
#include <hip/hip_runtime.h>
#include <hip/hip_bf16.h>
#include <math.h>

typedef __bf16 bf16_t;
typedef __attribute__((ext_vector_type(16))) __bf16 v16bf;
typedef __attribute__((ext_vector_type(8)))  float  v8f;

#define NTOK  32768     // B*D*H*W
#define C_    192
#define HID_  768
#define NWIN  128       // B * 64 windows
#define L_    256
#define NH_   6
#define HD_   32
#define NB_   1575
#define SCALE_ 0.17677669529663687f   // 32^-0.5

// ---------------- WMMA fragment helpers (wave32, contiguous k order used by both A and B) -----

// A fragment 16x32 bf16, memory row-major [16 x ld]:
// lane l: m = l&15, half = l>>4; element e -> k = 16*half + e (contiguous)
__device__ inline v16bf frag_a_rowmajor(const bf16_t* base, int ld, int lane) {
  int m = lane & 15, half = lane >> 4;
  const bf16_t* row = base + m * ld + half * 16;
  v16bf f;
#pragma unroll
  for (int e = 0; e < 16; ++e) f[e] = row[e];
  return f;
}

// B fragment 32x16 bf16 where memory holds element (k,n) at base[n*ld + k]:
// lane l: n = l&15, half = l>>4; element e -> k = 16*half + e (contiguous!)
__device__ inline v16bf frag_b_kmajor(const bf16_t* base, int ld, int lane) {
  int n = lane & 15, half = lane >> 4;
  const bf16_t* p = base + n * ld + half * 16;
  v16bf f;
#pragma unroll
  for (int e = 0; e < 16; ++e) f[e] = p[e];
  return f;
}

__device__ inline v8f wmma_bf16(v16bf a, v16bf b, v8f c) {
  return __builtin_amdgcn_wmma_f32_16x16x32_bf16(false, a, false, b,
                                                 (short)0, c, false, false);
}

// ---------------- layout shuffles -------------------------------------------

__global__ void k_transpose_in(const float* __restrict__ x, float* __restrict__ xt) {
  int idx = blockIdx.x * blockDim.x + threadIdx.x;
  if (idx >= 2 * C_ * 16384) return;
  int b = idx / (C_ * 16384);
  int rem = idx % (C_ * 16384);
  int c = rem / 16384, s = rem % 16384;
  xt[(b * 16384 + s) * C_ + c] = x[idx];
}

__global__ void k_transpose_out(const float* __restrict__ xt, float* __restrict__ out) {
  int idx = blockIdx.x * blockDim.x + threadIdx.x;
  if (idx >= 2 * C_ * 16384) return;
  int b = idx / (C_ * 16384);
  int rem = idx % (C_ * 16384);
  int c = rem / 16384, s = rem % 16384;
  out[idx] = xt[(b * 16384 + s) * C_ + c];
}

__global__ void k_cvt(const float* __restrict__ in, bf16_t* __restrict__ out, int n) {
  int i = blockIdx.x * blockDim.x + threadIdx.x;
  if (i < n) out[i] = (bf16_t)in[i];
}

// ---------------- LayerNorm (+ optional shift + window partition) -----------
// mode 0: natural order (LN2).  mode 1: window partition.  mode 2: window+shift.
__global__ __launch_bounds__(256) void k_ln(const float* __restrict__ xt,
                                            const float* __restrict__ w,
                                            const float* __restrict__ b,
                                            bf16_t* __restrict__ out, int mode) {
  int wave = threadIdx.x >> 5, lane = threadIdx.x & 31;
  int r = blockIdx.x * 8 + wave;     // destination row
  int t = r;                         // source token (natural order)
  if (mode != 0) {
    int win = r >> 8, tok = r & 255;
    int bb = win >> 6, wi = win & 63;
    int n0 = wi >> 4, n1 = (wi >> 2) & 3, n2 = wi & 3;
    int wd = tok >> 6, wh = (tok >> 3) & 7, ww = tok & 7;
    int d = n0 * 4 + wd, h = n1 * 8 + wh, wc = n2 * 8 + ww;
    if (mode == 2) { d = (d + 14) & 15; h = (h + 28) & 31; wc = (wc + 28) & 31; }
    t = ((bb * 16 + d) * 32 + h) * 32 + wc;
  }
  const float* xr = xt + (size_t)t * C_;
  float v[6]; float s = 0.f;
#pragma unroll
  for (int j = 0; j < 6; ++j) { v[j] = xr[j * 32 + lane]; s += v[j]; }
  for (int o = 16; o > 0; o >>= 1) s += __shfl_xor(s, o, 32);
  float mu = s * (1.0f / 192.0f);
  float qs = 0.f;
#pragma unroll
  for (int j = 0; j < 6; ++j) { float d0 = v[j] - mu; qs += d0 * d0; }
  for (int o = 16; o > 0; o >>= 1) qs += __shfl_xor(qs, o, 32);
  float rstd = rsqrtf(qs * (1.0f / 192.0f) + 1e-5f);
  bf16_t* orow = out + (size_t)r * C_;
#pragma unroll
  for (int j = 0; j < 6; ++j) {
    int c = j * 32 + lane;
    orow[c] = (bf16_t)((v[j] - mu) * rstd * w[c] + b[c]);
  }
}

// ---------------- relative-position-bias gather -----------------------------
__global__ void k_bias(const float* __restrict__ rpb, const int* __restrict__ rpi,
                       float* __restrict__ out) {
  int idx = blockIdx.x * blockDim.x + threadIdx.x;
  if (idx >= NH_ * L_ * L_) return;
  int head = idx >> 16, pos = idx & 65535;
  out[idx] = rpb[head * NB_ + rpi[pos]];
}

// ---------------- generic WMMA GEMM: out = A[M,K] * W[N,K]^T + bias ---------
// Each wave: 32M x 64N (2 A-frags + 4 B-frags -> 8 WMMAs per k-step).
// Block (8 waves): 256M x 64N.
enum { EPI_QKV = 0, EPI_GELU = 1, EPI_PROJ = 2, EPI_RES = 3 };

template <int EPI>
__global__ __launch_bounds__(256) void k_gemm(
    const bf16_t* __restrict__ A, const bf16_t* __restrict__ W,
    const float* __restrict__ bias, int K, int N,
    bf16_t* __restrict__ outb, float* __restrict__ outf,
    bf16_t* __restrict__ qo, bf16_t* __restrict__ ko, bf16_t* __restrict__ vo,
    int shifted) {
  int wave = threadIdx.x >> 5, lane = threadIdx.x & 31;
  int half = lane >> 4, nl = lane & 15;
  int m0 = blockIdx.y * 256 + wave * 32;   // two 16-row tiles
  int n0 = blockIdx.x * 64;
  v8f acc[2][4];
#pragma unroll
  for (int mi = 0; mi < 2; ++mi)
#pragma unroll
    for (int t = 0; t < 4; ++t) acc[mi][t] = (v8f){};
  const bf16_t* aptr0 = A + (size_t)m0 * K;
  const bf16_t* aptr1 = A + (size_t)(m0 + 16) * K;
  for (int kk = 0; kk < K; kk += 32) {
    if (kk + 32 < K) {
      __builtin_prefetch(aptr0 + kk + 32, 0, 1);
      __builtin_prefetch(aptr1 + kk + 32, 0, 1);
    }
    v16bf a0 = frag_a_rowmajor(aptr0 + kk, K, lane);
    v16bf a1 = frag_a_rowmajor(aptr1 + kk, K, lane);
    v16bf b0 = frag_b_kmajor(W + (size_t)(n0 +  0) * K + kk, K, lane);
    v16bf b1 = frag_b_kmajor(W + (size_t)(n0 + 16) * K + kk, K, lane);
    v16bf b2 = frag_b_kmajor(W + (size_t)(n0 + 32) * K + kk, K, lane);
    v16bf b3 = frag_b_kmajor(W + (size_t)(n0 + 48) * K + kk, K, lane);
    acc[0][0] = wmma_bf16(a0, b0, acc[0][0]);
    acc[0][1] = wmma_bf16(a0, b1, acc[0][1]);
    acc[0][2] = wmma_bf16(a0, b2, acc[0][2]);
    acc[0][3] = wmma_bf16(a0, b3, acc[0][3]);
    acc[1][0] = wmma_bf16(a1, b0, acc[1][0]);
    acc[1][1] = wmma_bf16(a1, b1, acc[1][1]);
    acc[1][2] = wmma_bf16(a1, b2, acc[1][2]);
    acc[1][3] = wmma_bf16(a1, b3, acc[1][3]);
  }
#pragma unroll
  for (int mi = 0; mi < 2; ++mi) {
#pragma unroll
    for (int t = 0; t < 4; ++t) {
#pragma unroll
      for (int r = 0; r < 8; ++r) {
        int row = m0 + mi * 16 + half * 8 + r;
        int col = n0 + t * 16 + nl;
        float val = acc[mi][t][r] + bias[col];
        if (EPI == EPI_QKV) {
          int part = col / 192, rem = col % 192;
          int head = rem >> 5, hd = rem & 31;
          int win = row >> 8, tok = row & 255;
          size_t wh = (size_t)win * NH_ + head;
          if (part == 0)      qo[(wh * 256 + tok) * 32 + hd] = (bf16_t)(val * SCALE_);
          else if (part == 1) ko[(wh * 256 + tok) * 32 + hd] = (bf16_t)val;
          else                vo[(wh * 32 + hd) * 256 + tok] = (bf16_t)val;   // V^T
        } else if (EPI == EPI_GELU) {
          float g = 0.5f * val * (1.0f + erff(val * 0.70710678118654752f));
          outb[(size_t)row * N + col] = (bf16_t)g;
        } else if (EPI == EPI_PROJ) {
          int win = row >> 8, tok = row & 255;
          int bb = win >> 6, wi = win & 63;
          int nn0 = wi >> 4, nn1 = (wi >> 2) & 3, nn2 = wi & 3;
          int wd = tok >> 6, whh = (tok >> 3) & 7, ww = tok & 7;
          int d = nn0 * 4 + wd, h = nn1 * 8 + whh, wc = nn2 * 8 + ww;
          if (shifted) { d = (d + 14) & 15; h = (h + 28) & 31; wc = (wc + 28) & 31; }
          int tn = ((bb * 16 + d) * 32 + h) * 32 + wc;
          outf[(size_t)tn * C_ + col] += val;     // residual add, un-windowed
        } else {                                   // EPI_RES: natural order
          outf[(size_t)row * C_ + col] += val;
        }
      }
    }
  }
}

// ---------------- flash attention per (window, head) ------------------------
__global__ __launch_bounds__(512) void k_attn(
    const bf16_t* __restrict__ q, const bf16_t* __restrict__ kmat,
    const bf16_t* __restrict__ vt, const float* __restrict__ bias_full,
    const unsigned char* __restrict__ mask, bf16_t* __restrict__ out,
    int shifted) {
  __shared__ bf16_t lds_p[16 * 16 * 32];   // 16 waves x (16x32 bf16 P chunk)
  int wh = blockIdx.x;                     // win*NH + head
  int win = wh / NH_, head = wh % NH_;
  int wave = threadIdx.x >> 5, lane = threadIdx.x & 31;
  int half = lane >> 4, nl = lane & 15;
  int m0 = wave * 16;
  const bf16_t* qb = q    + (size_t)wh * (L_ * HD_);
  const bf16_t* kb = kmat + (size_t)wh * (L_ * HD_);
  const bf16_t* vb = vt   + (size_t)wh * (HD_ * L_);
  const float* bb = bias_full + (size_t)head * (L_ * L_);
  const unsigned char* mb = mask + (size_t)(win & 63) * (L_ * L_);

  v16bf aq = frag_a_rowmajor(qb + m0 * HD_, HD_, lane);  // Q 16x32 (pre-scaled)
  v8f o0 = {}, o1 = {};
  float mrun[8], lrun[8];
#pragma unroll
  for (int r = 0; r < 8; ++r) { mrun[r] = -1e30f; lrun[r] = 0.f; }
  bf16_t* myp = lds_p + wave * (16 * 32);

  for (int kc = 0; kc < 8; ++kc) {         // 32 keys per iteration
    int c0 = kc * 32;
    v16bf bk0 = frag_b_kmajor(kb + (size_t)(c0)      * HD_, HD_, lane);
    v16bf bk1 = frag_b_kmajor(kb + (size_t)(c0 + 16) * HD_, HD_, lane);
    v8f z = {};
    v8f s0 = wmma_bf16(aq, bk0, z);
    v8f s1 = wmma_bf16(aq, bk1, z);
#pragma unroll
    for (int r = 0; r < 8; ++r) {
      int row = m0 + half * 8 + r;
      int col0 = c0 + nl, col1 = c0 + 16 + nl;
      float v0 = s0[r] + bb[row * L_ + col0];
      float v1 = s1[r] + bb[row * L_ + col1];
      if (shifted) {
        if (!mb[row * L_ + col0]) v0 = -1e30f;
        if (!mb[row * L_ + col1]) v1 = -1e30f;
      }
      float tmx = fmaxf(v0, v1);
      for (int o = 8; o > 0; o >>= 1) tmx = fmaxf(tmx, __shfl_xor(tmx, o, 32));
      float mnew = fmaxf(mrun[r], tmx);
      float alpha = __expf(mrun[r] - mnew);
      mrun[r] = mnew;
      float p0 = __expf(v0 - mnew);
      float p1 = __expf(v1 - mnew);
      float rs = p0 + p1;
      for (int o = 8; o > 0; o >>= 1) rs += __shfl_xor(rs, o, 32);
      lrun[r] = lrun[r] * alpha + rs;
      o0[r] *= alpha;
      o1[r] *= alpha;
      myp[(half * 8 + r) * 32 + nl]      = (bf16_t)p0;   // stage P row-major
      myp[(half * 8 + r) * 32 + 16 + nl] = (bf16_t)p1;
    }
    __syncthreads();
    v16bf ap  = frag_a_rowmajor(myp, 32, lane);          // P 16x32 from LDS
    v16bf bv0 = frag_b_kmajor(vb + c0,            L_, lane);  // V^T rows hd 0..15
    v16bf bv1 = frag_b_kmajor(vb + 16 * L_ + c0,  L_, lane);  // hd 16..31
    o0 = wmma_bf16(ap, bv0, o0);
    o1 = wmma_bf16(ap, bv1, o1);
    __syncthreads();
  }
#pragma unroll
  for (int r = 0; r < 8; ++r) {
    int tok = m0 + half * 8 + r;
    float inv = 1.0f / lrun[r];
    size_t base = ((size_t)win * 256 + tok) * C_ + head * 32;
    out[base + nl]      = (bf16_t)(o0[r] * inv);
    out[base + 16 + nl] = (bf16_t)(o1[r] * inv);
  }
}

// ---------------- launcher --------------------------------------------------
extern "C" void kernel_launch(void* const* d_in, const int* in_sizes, int n_in,
                              void* d_out, int out_size, void* d_ws, size_t ws_size,
                              hipStream_t stream) {
  (void)in_sizes; (void)n_in; (void)out_size; (void)ws_size;
  const float* x      = (const float*)d_in[0];
  const float* ln1_w  = (const float*)d_in[1];
  const float* ln1_b  = (const float*)d_in[2];
  const float* qkv_w  = (const float*)d_in[3];
  const float* qkv_b  = (const float*)d_in[4];
  const float* rpb    = (const float*)d_in[5];
  const float* proj_w = (const float*)d_in[6];
  const float* proj_b = (const float*)d_in[7];
  const float* ln2_w  = (const float*)d_in[8];
  const float* ln2_b  = (const float*)d_in[9];
  const float* fc1_w  = (const float*)d_in[10];
  const float* fc1_b  = (const float*)d_in[11];
  const float* fc2_w  = (const float*)d_in[12];
  const float* fc2_b  = (const float*)d_in[13];
  const int*   rpi    = (const int*)d_in[14];
  const unsigned char* amask = (const unsigned char*)d_in[15];

  char* ws = (char*)d_ws;
  size_t off = 0;
  auto alloc = [&](size_t bytes) -> void* {
    void* p = ws + off; off += (bytes + 255) & ~(size_t)255; return p;
  };
  float*  xt   = (float*) alloc((size_t)NTOK * C_ * 4);   // residual stream
  bf16_t* bufA = (bf16_t*)alloc((size_t)NTOK * C_ * 2);   // LN outputs / GEMM A
  bf16_t* bufO = (bf16_t*)alloc((size_t)NTOK * C_ * 2);   // attention output
  float*  bfl  = (float*) alloc((size_t)NH_ * L_ * L_ * 4);
  bf16_t* wbuf = (bf16_t*)alloc((size_t)HID_ * C_ * 2);   // converted weights
  char*   un   = (char*)  alloc((size_t)NTOK * HID_ * 2); // union: q/k/vT | mid
  bf16_t* qb  = (bf16_t*)un;
  bf16_t* kb  = (bf16_t*)(un + (size_t)NTOK * C_ * 2);
  bf16_t* vb  = (bf16_t*)(un + 2 * (size_t)NTOK * C_ * 2);
  bf16_t* mid = (bf16_t*)un;

  k_transpose_in<<<(NTOK * C_ + 255) / 256, 256, 0, stream>>>(x, xt);

  for (int i = 0; i < 2; ++i) {
    int shifted = i & 1;
    // LN1 + (shift) + window partition -> bufA
    k_ln<<<NTOK / 8, 256, 0, stream>>>(xt, ln1_w + i * C_, ln1_b + i * C_,
                                       bufA, shifted ? 2 : 1);
    // QKV
    k_cvt<<<(576 * C_ + 255) / 256, 256, 0, stream>>>(qkv_w + (size_t)i * 576 * C_,
                                                      wbuf, 576 * C_);
    k_gemm<EPI_QKV><<<dim3(576 / 64, NTOK / 256), 256, 0, stream>>>(
        bufA, wbuf, qkv_b + i * 576, C_, 576,
        nullptr, nullptr, qb, kb, vb, 0);
    // bias table
    k_bias<<<(NH_ * L_ * L_ + 255) / 256, 256, 0, stream>>>(
        rpb + (size_t)i * NH_ * NB_, rpi, bfl);
    // attention
    k_attn<<<NWIN * NH_, 512, 0, stream>>>(qb, kb, vb, bfl, amask, bufO, shifted);
    // proj + un-window/un-shift + residual
    k_cvt<<<(C_ * C_ + 255) / 256, 256, 0, stream>>>(proj_w + (size_t)i * C_ * C_,
                                                     wbuf, C_ * C_);
    k_gemm<EPI_PROJ><<<dim3(C_ / 64, NTOK / 256), 256, 0, stream>>>(
        bufO, wbuf, proj_b + i * C_, C_, C_,
        nullptr, xt, nullptr, nullptr, nullptr, shifted);
    // LN2 -> bufA
    k_ln<<<NTOK / 8, 256, 0, stream>>>(xt, ln2_w + i * C_, ln2_b + i * C_, bufA, 0);
    // fc1 + GELU
    k_cvt<<<(HID_ * C_ + 255) / 256, 256, 0, stream>>>(fc1_w + (size_t)i * HID_ * C_,
                                                       wbuf, HID_ * C_);
    k_gemm<EPI_GELU><<<dim3(HID_ / 64, NTOK / 256), 256, 0, stream>>>(
        bufA, wbuf, fc1_b + i * HID_, C_, HID_,
        mid, nullptr, nullptr, nullptr, nullptr, 0);
    // fc2 + residual
    k_cvt<<<(C_ * HID_ + 255) / 256, 256, 0, stream>>>(fc2_w + (size_t)i * C_ * HID_,
                                                       wbuf, C_ * HID_);
    k_gemm<EPI_RES><<<dim3(C_ / 64, NTOK / 256), 256, 0, stream>>>(
        mid, wbuf, fc2_b + i * C_, HID_, C_,
        nullptr, xt, nullptr, nullptr, nullptr, 0);
  }

  k_transpose_out<<<(NTOK * C_ + 255) / 256, 256, 0, stream>>>(xt, (float*)d_out);
}
